// ArgumentGCN_39015482917621
// MI455X (gfx1250) — matrix operand order repaired and checked by reading, not compiled, not run on hardware
//
#include <hip/hip_runtime.h>
#include <hip/hip_bf16.h>
#include <math.h>

#define Bb 8
#define Nn 2048
#define Dd 128
#define STEPS 2
#define NIT (Nn / 32)          // k-chunks in aggregation

typedef __attribute__((ext_vector_type(16))) __bf16          v16bf;
typedef __attribute__((ext_vector_type(8)))  float           v8f;
typedef __attribute__((ext_vector_type(4)))  float           v4f;
typedef __attribute__((ext_vector_type(4)))  int             v4i;
typedef __attribute__((ext_vector_type(8)))  int             v8i;
typedef __attribute__((ext_vector_type(4)))  unsigned int    v4ui;
typedef __attribute__((ext_vector_type(8)))  unsigned short  v8us;

__device__ __forceinline__ unsigned short bf16_bits(float f) {
    __bf16 h = (__bf16)f;                   // native v_cvt f32->bf16
    unsigned short u; __builtin_memcpy(&u, &h, 2); return u;
}
__device__ __forceinline__ __bf16 bits_to_bf16(unsigned short u) {
    __bf16 b; __builtin_memcpy(&b, &u, 2); return b;
}

// ---------------------------------------------------------------------------
// TDM: issue a 2-D tensor tile load (tile_dim0=32 elems along j, tile_dim1=128
// rows, data_size=2B, tensor stride 2048) from global bf16 into LDS (packed,
// 64-B rows).  D# per CDNA5 ISA §8.3/8.4.  Wave-level op; tracked by TENSORcnt.
// 6-arg builtin form (clang-23 / therock-10.0 toolchain).
// ---------------------------------------------------------------------------
__device__ __forceinline__ void tdm_load_chunk(unsigned long long gaddr,
                                               unsigned lds_addr) {
    v4ui g0;
    g0[0] = 1u;                                   // count=1, user descriptor
    g0[1] = lds_addr;                             // lds_addr (bytes)
    g0[2] = (unsigned)(gaddr & 0xFFFFFFFFull);    // global_addr[31:0]
    g0[3] = (unsigned)((gaddr >> 32) & 0x1FFFFFFull) | (2u << 30); // [56:32] + type=2
    v8i g1;
    g1[0] = 1 << 16;                              // data_size=1 (2 bytes)
    g1[1] = (int)(2048u << 16);                   // tensor_dim0 = 2048 (bits 79:48)
    g1[2] = (int)(128u  << 16);                   // tensor_dim1 = 128  (bits 111:80)
    g1[3] = (int)(32u   << 16);                   // tile_dim0  = 32   (bits 127:112)
    g1[4] = 128;                                  // tile_dim1 = 128, tile_dim2 = 0
    g1[5] = 2048;                                 // tensor_dim0_stride = 2048
    g1[6] = 0;                                    // stride0 hi / stride1 lo
    g1[7] = 0;                                    // stride1 hi
    v4i g2 = (v4i){0, 0, 0, 0};
    v4i g3 = (v4i){0, 0, 0, 0};
    v8i g4 = (v8i){0, 0, 0, 0, 0, 0, 0, 0};
    __builtin_amdgcn_tensor_load_to_lds(g0, g1, g2, g3, g4, 0);
}

// ---------------------------------------------------------------------------
// nbr_inv[b*N+i] = 1 / max(1, sum_j mask_i*mask_j*(i!=j)*punct[b,i,j])
// ---------------------------------------------------------------------------
__global__ void __launch_bounds__(256) k_nbr_inv(const int* __restrict__ mask,
                                                 const int* __restrict__ punct,
                                                 float* __restrict__ nbr_inv) {
    __shared__ int red[256];
    int row = blockIdx.x;                 // b*N + i
    int b = row >> 11;
    int i = row & (Nn - 1);
    const int* prow = punct + (long long)row * Nn;
    const int* mrow = mask + b * Nn;
    int mi = mrow[i];
    int s = 0;
    for (int j = threadIdx.x; j < Nn; j += 256) {
        int v = (prow[j] != 0 && mrow[j] != 0) ? 1 : 0;
        if (j == i) v = 0;
        s += v;
    }
    red[threadIdx.x] = s;
    __syncthreads();
    for (int off = 128; off > 0; off >>= 1) {
        if (threadIdx.x < (unsigned)off) red[threadIdx.x] += red[threadIdx.x + off];
        __syncthreads();
    }
    if (threadIdx.x == 0) {
        int tot = (mi != 0) ? red[0] : 0;
        nbr_inv[row] = 1.0f / (float)(tot >= 1 ? tot : 1);
    }
}

// ---------------------------------------------------------------------------
// dw[b*N+n] = sigmoid(x[b,n,:]·w_nw + b_nw); also writes all_weight[b,step,n].
// ---------------------------------------------------------------------------
__global__ void __launch_bounds__(256) k_dw(const float* __restrict__ x,
                                            const float* __restrict__ w_nw,
                                            const float* __restrict__ b_nw,
                                            float* __restrict__ dw,
                                            float* __restrict__ allw,
                                            int step) {
    int wid = threadIdx.x >> 5, lane = threadIdx.x & 31;
    int row = blockIdx.x * 8 + wid;        // b*N + n
    const float* xr = x + (long long)row * Dd;
    float s = 0.f;
    #pragma unroll
    for (int k = lane; k < Dd; k += 32) s += xr[k] * w_nw[k];
    #pragma unroll
    for (int off = 16; off > 0; off >>= 1) s += __shfl_xor(s, off, 32);
    if (lane == 0) {
        float z  = s + b_nw[0];
        float sg = 1.0f / (1.0f + __expf(-z));
        dw[row] = sg;
        int b = row >> 11, n = row & (Nn - 1);
        allw[((long long)b * STEPS + step) * Nn + n] = sg;
    }
}

// ---------------------------------------------------------------------------
// out[m,e] = sum_d x[m,d] * W[e,d] (+ bias[e]) — M=16384, 128x128 weights.
// mode 0: float output + bias (self_info).
// mode 1: bf16 output TRANSPOSED to [b][e][j] via LDS tile (info_p for TDM).
// ---------------------------------------------------------------------------
__global__ void __launch_bounds__(256) k_gemm_xw(const float* __restrict__ x,
                                                 const float* __restrict__ W,
                                                 const float* __restrict__ bias,
                                                 float* __restrict__ outF,
                                                 unsigned short* __restrict__ outT,
                                                 int mode) {
    __shared__ unsigned short Wl[Dd * Dd];        // 32 KB bf16, row-major [e][d]
    __shared__ unsigned short Tl[Dd * Dd];        // 32 KB transpose tile (mode 1)
    for (int t = threadIdx.x; t < Dd * Dd; t += 256) Wl[t] = bf16_bits(W[t]);
    __syncthreads();

    int wid = threadIdx.x >> 5, lane = threadIdx.x & 31;
    int row0 = (blockIdx.x * 8 + wid) * 16;
    int m = lane & 15, kh = lane >> 4;
    const float* xr = x + (long long)(row0 + m) * Dd;

    v8f acc[8];
    #pragma unroll
    for (int t = 0; t < 8; ++t) acc[t] = (v8f){0.f,0.f,0.f,0.f,0.f,0.f,0.f,0.f};

    #pragma unroll
    for (int kk = 0; kk < Dd; kk += 32) {
        // Two contiguous 8-float runs per lane: k = kk+8*kh..+7 and kk+16+8*kh..+7
        v4f x0 = *(const v4f*)(xr + kk + (kh << 3));
        v4f x1 = *(const v4f*)(xr + kk + (kh << 3) + 4);
        v4f x2 = *(const v4f*)(xr + kk + 16 + (kh << 3));
        v4f x3 = *(const v4f*)(xr + kk + 16 + (kh << 3) + 4);
        float run[16];
        #pragma unroll
        for (int e = 0; e < 4; ++e) {
            run[e] = x0[e]; run[4 + e] = x1[e]; run[8 + e] = x2[e]; run[12 + e] = x3[e];
        }
        v16bf A;   // ISA 16-bit A layout: pos p = (v>=4 ? 8 : 0) + 2*(v&3)
        #pragma unroll
        for (int v = 0; v < 8; ++v) {
            int p = ((v & 4) << 1) + ((v & 3) << 1);
            A[2 * v]     = (__bf16)run[p];
            A[2 * v + 1] = (__bf16)run[p + 1];
        }
        #pragma unroll
        for (int t = 0; t < 8; ++t) {
            int col = (t << 4) + m;              // output dim 'e'
            v16bf Bf;                            // B[k][col] = W[col, k]; contiguous in e
            #pragma unroll
            for (int e = 0; e < 16; ++e)
                Bf[e] = bits_to_bf16(Wl[col * Dd + kk + (kh << 3) + (e & 7) + ((e & 8) << 1)]);
            acc[t] = __builtin_amdgcn_wmma_f32_16x16x32_bf16(
                false, A, false, Bf, (short)0, acc[t], false, false);
        }
    }

    if (mode == 0) {
        #pragma unroll
        for (int t = 0; t < 8; ++t) {
            int colg = (t << 4) + m;
            float bv = bias[colg];
            #pragma unroll
            for (int r = 0; r < 8; ++r) {
                int rm = r + (kh << 3);
                outF[(long long)(row0 + rm) * Dd + colg] = acc[t][r] + bv;
            }
        }
    } else {
        // stage bf16 accums into LDS [c][row_local], then coalesced transposed store
        int rbase = wid << 4;
        #pragma unroll
        for (int t = 0; t < 8; ++t) {
            int colg = (t << 4) + m;
            #pragma unroll
            for (int r = 0; r < 8; ++r)
                Tl[colg * Dd + rbase + r + (kh << 3)] = bf16_bits(acc[t][r]);
        }
        __syncthreads();
        int row0b = blockIdx.x * 128;
        int bb = row0b >> 11;                 // batch
        int j0 = row0b & (Nn - 1);            // node offset within batch
        for (int t = threadIdx.x; t < Dd * 16; t += 256) {
            int c = t >> 4, ch = t & 15;      // 16 chunks of 8 bf16 per feature row
            *(v8us*)(outT + ((size_t)bb * Dd + c) * Nn + j0 + (ch << 3)) =
                *(const v8us*)&Tl[c * Dd + (ch << 3)];
        }
    }
}

// ---------------------------------------------------------------------------
// agg[b,i,:] = ( sum_j NW[b,i,j] * info_p[b,j,:] ) * nbr_inv[b,i]
// x_out = relu(self_info + agg)
// info_p held transposed bf16 [b][c][j]; k-chunks DMA'd into LDS by the
// Tensor Data Mover (double-buffered), B frags = 2x ds_load_b128.
// ---------------------------------------------------------------------------
__global__ void __launch_bounds__(256) k_agg(const unsigned short* __restrict__ ipT,
                                             const float* __restrict__ self_info,
                                             const int*   __restrict__ punct,
                                             const int*   __restrict__ mask,
                                             const float* __restrict__ dw,
                                             const float* __restrict__ nbr_inv,
                                             float* __restrict__ xout) {
    __shared__ __align__(16) unsigned short Bl[2][32 * Dd]; // 2 x 8 KB TDM buffers [c][k]
    __shared__ __align__(16) unsigned short Avb[32];        // bf16(mask_j ? dw[j] : 0)

    int wid = threadIdx.x >> 5, lane = threadIdx.x & 31;
    int b  = blockIdx.x >> 4;
    int i0 = ((blockIdx.x & 15) * 8 + wid) * 16;
    int m  = lane & 15, kh = lane >> 4;
    int gi = i0 + m;                         // row within batch for A loads
    const int*   prow = punct + ((long long)b * Nn + gi) * Nn;
    const int*   mrow = mask + b * Nn;
    const float* dwb  = dw + (long long)b * Nn;
    int mi = mrow[gi];

    unsigned long long gbase =
        (unsigned long long)(size_t)(ipT + (size_t)b * Dd * Nn);
    unsigned lds0 = (unsigned)(size_t)(void*)&Bl[0][0];
    unsigned lds1 = (unsigned)(size_t)(void*)&Bl[1][0];

    v8f acc[8];
    #pragma unroll
    for (int t = 0; t < 8; ++t) acc[t] = (v8f){0.f,0.f,0.f,0.f,0.f,0.f,0.f,0.f};

    if (wid == 0) tdm_load_chunk(gbase, lds0);            // prologue: chunk 0

    for (int it = 0; it < NIT; ++it) {
        int k0 = it << 5;
        if (wid == 0) {
            if (it + 1 < NIT) {
                tdm_load_chunk(gbase + (unsigned long long)(k0 + 32) * 2ull,
                               ((it + 1) & 1) ? lds1 : lds0);
                __builtin_amdgcn_s_wait_tensorcnt(1);     // chunk `it` landed
            } else {
                __builtin_amdgcn_s_wait_tensorcnt(0);
            }
        }
        if (threadIdx.x < 32) {
            int j = k0 + threadIdx.x;
            Avb[threadIdx.x] = bf16_bits((mrow[j] != 0) ? dwb[j] : 0.0f);
        }
        __syncthreads();                                  // publish chunk + Avb

        // A fragment = NW tile [16 x 32]: two 8-wide runs per lane half
        int jlo = k0 + (kh << 3);
        int jhi = jlo + 16;
        v4i p0 = *(const v4i*)(prow + jlo);
        v4i p1 = *(const v4i*)(prow + jlo + 4);
        v4i p2 = *(const v4i*)(prow + jhi);
        v4i p3 = *(const v4i*)(prow + jhi + 4);
        v8us alo = *(const v8us*)&Avb[kh << 3];
        v8us ahi = *(const v8us*)&Avb[16 + (kh << 3)];
        int pr[16]; unsigned short av[16]; int jg[16];
        #pragma unroll
        for (int e = 0; e < 4; ++e) {
            pr[e]      = p0[e]; pr[4 + e]  = p1[e];
            pr[8 + e]  = p2[e]; pr[12 + e] = p3[e];
        }
        #pragma unroll
        for (int e = 0; e < 8; ++e) {
            av[e] = alo[e]; av[8 + e] = ahi[e];
            jg[e] = jlo + e; jg[8 + e] = jhi + e;
        }
        v16bf A;
        #pragma unroll
        for (int v = 0; v < 8; ++v) {
            int p = ((v & 4) << 1) + ((v & 3) << 1);
            unsigned short a0 = (mi != 0 && pr[p]     != 0 && jg[p]     != gi) ? av[p]     : (unsigned short)0;
            unsigned short a1 = (mi != 0 && pr[p + 1] != 0 && jg[p + 1] != gi) ? av[p + 1] : (unsigned short)0;
            A[2 * v]     = bits_to_bf16(a0);
            A[2 * v + 1] = bits_to_bf16(a1);
        }

        const unsigned short* Bc = &Bl[it & 1][0];        // [c][k] rows of 32, 64 B
        #pragma unroll
        for (int t = 0; t < 8; ++t) {
            int col = (t << 4) + m;
            v16bf Bf;                                     // contiguous in k
            #pragma unroll
            for (int e = 0; e < 16; ++e)
                Bf[e] = bits_to_bf16(Bc[(col << 5) + (kh << 3) + (e & 7) + ((e & 8) << 1)]);
            acc[t] = __builtin_amdgcn_wmma_f32_16x16x32_bf16(
                false, A, false, Bf, (short)0, acc[t], false, false);
        }
        __syncthreads();                                  // done reading this buffer
    }

    float invn[8];
    #pragma unroll
    for (int r = 0; r < 8; ++r)
        invn[r] = nbr_inv[(long long)b * Nn + i0 + r + (kh << 3)];

    #pragma unroll
    for (int t = 0; t < 8; ++t) {
        int colg = (t << 4) + m;
        #pragma unroll
        for (int r = 0; r < 8; ++r) {
            int rm = r + (kh << 3);
            long long rowg = (long long)b * Nn + i0 + rm;
            float v = self_info[rowg * Dd + colg] + acc[t][r] * invn[r];
            xout[rowg * Dd + colg] = v > 0.f ? v : 0.f;
        }
    }
}

// ---------------------------------------------------------------------------
extern "C" void kernel_launch(void* const* d_in, const int* in_sizes, int n_in,
                              void* d_out, int out_size, void* d_ws, size_t ws_size,
                              hipStream_t stream) {
    const float* node      = (const float*)d_in[0];   // [B,N,D]
    const int*   node_mask = (const int*)  d_in[1];   // [B,N]
    const int*   punct     = (const int*)  d_in[2];   // [B,N,N]
    const float* w_nw      = (const float*)d_in[3];   // [1,D]
    const float* b_nw      = (const float*)d_in[4];   // [1]
    const float* w_self    = (const float*)d_in[5];   // [D,D]
    const float* b_self    = (const float*)d_in[6];   // [D]
    const float* w_punct   = (const float*)d_in[7];   // [D,D]
    (void)in_sizes; (void)n_in; (void)out_size; (void)ws_size;

    const size_t XSZ = (size_t)Bb * Nn * Dd;
    float* ws        = (float*)d_ws;
    float* x1        = ws;                            // XSZ floats
    float* self_info = x1 + XSZ;                      // XSZ floats
    float* dwbuf     = self_info + XSZ;               // B*N floats
    float* nbr_inv   = dwbuf + (size_t)Bb * Nn;       // B*N floats
    unsigned short* ipT = (unsigned short*)(nbr_inv + (size_t)Bb * Nn); // XSZ bf16

    float* x_out = (float*)d_out;                     // [B,N,D]
    float* allw  = x_out + XSZ;                       // [B,STEPS,N]

    k_nbr_inv<<<Bb * Nn, 256, 0, stream>>>(node_mask, punct, nbr_inv);

    const float* xc = node;
    for (int step = 0; step < STEPS; ++step) {
        k_dw<<<(Bb * Nn) / 8, 256, 0, stream>>>(xc, w_nw, b_nw, dwbuf, allw, step);
        k_gemm_xw<<<(Bb * Nn) / 128, 256, 0, stream>>>(xc, w_self, b_self,
                                                       self_info, nullptr, 0);
        k_gemm_xw<<<(Bb * Nn) / 128, 256, 0, stream>>>(xc, w_punct, nullptr,
                                                       nullptr, ipT, 1);
        float* xn = (step == STEPS - 1) ? x_out : x1;
        k_agg<<<Bb * 16, 256, 0, stream>>>(ipT, self_info, punct, node_mask,
                                           dwbuf, nbr_inv, xn);
        xc = xn;
    }
}
